// SCAESuite_65420941853531
// MI455X (gfx1250) — matrix operand order repaired and checked by reading, not compiled, not run on hardware
//
#include <hip/hip_runtime.h>

#define B_SZ   4096
#define D_SZ   768
#define FUP    12288
#define FDOWN  12288
#define C_SZ   256
#define K_TOP  64
#define KSTAGE 64                   // K elements staged per TDM tile
#define LDSTR  72                   // LDS row stride in bf16 elems (144B, HW-padded)

typedef __attribute__((ext_vector_type(16))) __bf16    v16bf;
typedef __attribute__((ext_vector_type(8)))  float     v8f;
typedef __attribute__((ext_vector_type(4)))  unsigned  v4u;
typedef __attribute__((ext_vector_type(8)))  int       v8i_t;
typedef __attribute__((ext_vector_type(4)))  int       v4i_t;

#if defined(__has_builtin)
#  if __has_builtin(__builtin_amdgcn_tensor_load_to_lds)
#    define HAS_TDM 1
#  else
#    define HAS_TDM 0
#  endif
#else
#  define HAS_TDM 0
#endif

__device__ __forceinline__ __bf16 f2bf(float f) {
    unsigned u = __float_as_uint(f);
    u += 0x7FFFu + ((u >> 16) & 1u);          // round-to-nearest-even
    unsigned short h = (unsigned short)(u >> 16);
    return __builtin_bit_cast(__bf16, h);
}

// ---------------------------------------------------------------------------
// TDM: async 2D bf16 tile load Global -> LDS.
// Tile row = 64 bf16 = 128B; pad_interval=4 (pad after every 32 DWORDs=128B),
// pad_amount=3 (4 DWORDs=16B)  ==>  LDS row stride 144B == [rows][72] bf16.
// ---------------------------------------------------------------------------
__device__ __forceinline__ void tdm_load_tile_bf16(
    unsigned lds_addr, const __bf16* gptr,
    unsigned tile_x, unsigned tile_y,
    unsigned tensor_x, unsigned tensor_y, unsigned stride_x /*elems*/)
{
#if HAS_TDM
    unsigned long long ga = (unsigned long long)(size_t)gptr;
    v4u g0;
    g0[0] = 1u;                                            // count=1 (valid, user)
    g0[1] = lds_addr;                                      // LDS byte address
    g0[2] = (unsigned)ga;                                  // global_addr[31:0]
    g0[3] = (unsigned)((ga >> 32) & 0x01FFFFFFu) | (2u << 30);  // addr[56:32] | type=2
    v8i_t g1;
    g1[0] = (int)((1u << 16)        // data_size = 1 (2 bytes)
                | (1u << 20)        // pad_enable
                | (4u << 22)        // pad_interval: 32 DWORDs (128B)
                | (3u << 25));      // pad_amount:   4 DWORDs (16B)
    g1[1] = (int)((tensor_x & 0xFFFFu) << 16);                       // dim0[15:0]
    g1[2] = (int)((tensor_x >> 16) | ((tensor_y & 0xFFFFu) << 16));  // dim0 hi | dim1 lo
    g1[3] = (int)((tensor_y >> 16) | (tile_x << 16));                // dim1 hi | tile_dim0
    g1[4] = (int)(tile_y & 0xFFFFu);                                 // tile_dim1, tile_dim2=0
    g1[5] = (int)stride_x;                                           // dim0_stride[31:0]
    g1[6] = 0;
    g1[7] = 0;
    v4i_t z4{};
#  if __has_include(<hip/amd_detail/amd_gfx1250_TDM.h>)
    v8i_t z8{};
    __builtin_amdgcn_tensor_load_to_lds(g0, g1, z4, z4, z8, 0);
#  else
    __builtin_amdgcn_tensor_load_to_lds(g0, g1, z4, z4, 0);
#  endif
#else
    (void)lds_addr; (void)gptr; (void)tile_x; (void)tile_y;
    (void)tensor_x; (void)tensor_y; (void)stride_x;
#endif
}

// ---------------------------------------------------------------------------
// f32 -> bf16 conversion (optionally fused broadcast subtract of a row bias)
// ---------------------------------------------------------------------------
__global__ __launch_bounds__(256)
void cvt_bf16(const float* __restrict__ in, const float* __restrict__ sub,
              __bf16* __restrict__ out, int rowlen, long long n)
{
    long long i = (long long)blockIdx.x * 256 + threadIdx.x;
    if (i >= n) return;
    float v = in[i];
    if (sub) v -= sub[(int)(i % rowlen)];
    out[i] = f2bf(v);
}

// ---------------------------------------------------------------------------
// Dense GEMM:  C[M,N] = act( A[M,K] @ W[N,K]^T + bias ),  bf16 WMMA 16x16x32.
// 64x128 tile per 256-thread (8-wave) WG. TDM double-buffered LDS staging,
// 64-wide K stages; all fragments register-loaded before the 4-WMMA burst.
// ---------------------------------------------------------------------------
template <int RELU>
__global__ __launch_bounds__(256)
void gemm_nt_bf16(const __bf16* __restrict__ Abf, const __bf16* __restrict__ Wbf,
                  const float* __restrict__ bias, float* __restrict__ Cout,
                  int M, int N, int Kdim)
{
    __shared__ __bf16 As[2][64][LDSTR];
    __shared__ __bf16 Ws[2][128][LDSTR];

    const int tid  = threadIdx.x;
    const int lane = tid & 31;
    const int wave = tid >> 5;
    const int wm   = wave & 3;       // 4 row strips of 16
    const int wn   = wave >> 2;      // 2 col strips of 64
    const int m0   = blockIdx.y * 64;
    const int n0   = blockIdx.x * 128;
    const int nstage = Kdim / KSTAGE;

    v8f acc[4];
    for (int t = 0; t < 4; ++t)
        for (int i = 0; i < 8; ++i) acc[t][i] = 0.f;

#if HAS_TDM
    if (wave == 0) {
        tdm_load_tile_bf16((unsigned)(size_t)&As[0][0][0],
                           Abf + (size_t)m0 * Kdim, KSTAGE, 64, Kdim, M, Kdim);
        tdm_load_tile_bf16((unsigned)(size_t)&Ws[0][0][0],
                           Wbf + (size_t)n0 * Kdim, KSTAGE, 128, Kdim, N, Kdim);
    }
#endif

    for (int i = 0; i < nstage; ++i) {
        const int buf = i & 1;
        const int kk  = i * KSTAGE;
        __syncthreads();   // previous compute done: buf^1 is free to overwrite
#if HAS_TDM
        if (wave == 0) {
            if (i + 1 < nstage) {
                tdm_load_tile_bf16((unsigned)(size_t)&As[buf ^ 1][0][0],
                                   Abf + (size_t)m0 * Kdim + kk + KSTAGE,
                                   KSTAGE, 64, Kdim, M, Kdim);
                tdm_load_tile_bf16((unsigned)(size_t)&Ws[buf ^ 1][0][0],
                                   Wbf + (size_t)n0 * Kdim + kk + KSTAGE,
                                   KSTAGE, 128, Kdim, N, Kdim);
                __builtin_amdgcn_s_wait_tensorcnt(2);   // stage i done, i+1 in flight
            } else {
                __builtin_amdgcn_s_wait_tensorcnt(0);
            }
        }
#else
        for (int e = tid; e < 64 * KSTAGE; e += 256) {
            int r = e / KSTAGE, c = e % KSTAGE;
            As[buf][r][c] = Abf[(size_t)(m0 + r) * Kdim + kk + c];
        }
        for (int e = tid; e < 128 * KSTAGE; e += 256) {
            int r = e / KSTAGE, c = e % KSTAGE;
            Ws[buf][r][c] = Wbf[(size_t)(n0 + r) * Kdim + kk + c];
        }
#endif
        __syncthreads();

#pragma unroll
        for (int sub = 0; sub < KSTAGE / 32; ++sub) {
            const int ko = sub * 32;
            // A fragment: contiguous k = 16*(lane/16) + s (matches B)
            const int arow  = wm * 16 + (lane & 15);
            const int kbase = ko + ((lane < 16) ? 0 : 16);
            v16bf a;
#pragma unroll
            for (int v = 0; v < 4; ++v) {
                a[2 * v]     = As[buf][arow][kbase + 2 * v];
                a[2 * v + 1] = As[buf][arow][kbase + 2 * v + 1];
            }
#pragma unroll
            for (int v = 4; v < 8; ++v) {
                int k0 = kbase + 8 + 2 * (v - 4);
                a[2 * v]     = As[buf][arow][k0];
                a[2 * v + 1] = As[buf][arow][k0 + 1];
            }
            // all four B fragments before the WMMA burst
            const int kb = ko + ((lane < 16) ? 0 : 16);
            v16bf bfr[4];
#pragma unroll
            for (int t = 0; t < 4; ++t) {
                int bn = wn * 64 + t * 16 + (lane & 15);
#pragma unroll
                for (int v = 0; v < 8; ++v) {
                    bfr[t][2 * v]     = Ws[buf][bn][kb + 2 * v];
                    bfr[t][2 * v + 1] = Ws[buf][bn][kb + 2 * v + 1];
                }
            }
#pragma unroll
            for (int t = 0; t < 4; ++t)
                acc[t] = __builtin_amdgcn_wmma_f32_16x16x32_bf16(
                    false, a, false, bfr[t], (short)0, acc[t], false, false);
        }
    }

    // epilogue: VGPR r -> M=r (lanes 0-15), M=8+r (lanes 16-31)
#pragma unroll
    for (int t = 0; t < 4; ++t) {
        int n  = n0 + wn * 64 + t * 16 + (lane & 15);
        int mb = m0 + wm * 16 + ((lane < 16) ? 0 : 8);
        float bv = bias ? bias[n] : 0.f;
#pragma unroll
        for (int r = 0; r < 8; ++r) {
            float v = acc[t][r] + bv;
            if (RELU) v = fmaxf(v, 0.f);
            Cout[(size_t)(mb + r) * N + n] = v;
        }
    }
}

// ---------------------------------------------------------------------------
// Tiled transpose: in[R][Cc] -> out[Cc][R]
// ---------------------------------------------------------------------------
__global__ __launch_bounds__(256)
void transpose_kernel(const float* __restrict__ in, float* __restrict__ out,
                      int R, int Cc)
{
    __shared__ float t[32][33];
    int bx = blockIdx.x, by = blockIdx.y;
    int tx = threadIdx.x, ty = threadIdx.y;
    for (int i = ty; i < 32; i += 8)
        t[i][tx] = in[(size_t)(by * 32 + i) * Cc + bx * 32 + tx];
    __syncthreads();
    for (int i = ty; i < 32; i += 8)
        out[(size_t)(bx * 32 + i) * R + by * 32 + tx] = t[tx][i];
}

// ---------------------------------------------------------------------------
// CSC build of masked virtual weights: count / scan / fill
// ---------------------------------------------------------------------------
__global__ __launch_bounds__(256)
void count_kernel(const int* __restrict__ conn, int* __restrict__ counts)
{
    __shared__ int cj[C_SZ];
    int fd = blockIdx.x, c = threadIdx.x;
    cj[c] = conn[(size_t)fd * C_SZ + c];
    __syncthreads();
    int j = cj[c];
    bool dup = false;
    for (int c2 = 0; c2 < c; ++c2)
        if (cj[c2] == j) { dup = true; break; }
    if (!dup) atomicAdd(&counts[j], 1);
}

__global__ __launch_bounds__(1024)
void scan_kernel(const int* __restrict__ counts, int* __restrict__ colptr,
                 int* __restrict__ cursor)
{
    __shared__ int chunk[1024];
    const int CH = FUP / 1024;   // 12
    int tid = threadIdx.x;
    int s = 0;
    for (int i = 0; i < CH; ++i) s += counts[tid * CH + i];
    chunk[tid] = s;
    __syncthreads();
    if (tid == 0) {
        int acc = 0;
        for (int i = 0; i < 1024; ++i) { int t = chunk[i]; chunk[i] = acc; acc += t; }
        colptr[FUP] = acc;
    }
    __syncthreads();
    int run = chunk[tid];
    for (int i = 0; i < CH; ++i) {
        int j = tid * CH + i;
        colptr[j] = run; cursor[j] = run;
        run += counts[j];
    }
}

__global__ __launch_bounds__(256)
void fill_kernel(const int* __restrict__ conn, const float* __restrict__ W_enc_down,
                 const float* __restrict__ WduT, int* __restrict__ cursor,
                 int* __restrict__ col_fd, float* __restrict__ col_val)
{
    __shared__ int cj[C_SZ];
    __shared__ unsigned char dupf[C_SZ];
    __shared__ float wrow[D_SZ];
    int fd = blockIdx.x, tid = threadIdx.x;
    cj[tid] = conn[(size_t)fd * C_SZ + tid];
    for (int i = tid; i < D_SZ; i += 256) wrow[i] = W_enc_down[(size_t)fd * D_SZ + i];
    __syncthreads();
    {
        int j = cj[tid];
        bool d = false;
        for (int c2 = 0; c2 < tid; ++c2)
            if (cj[c2] == j) { d = true; break; }
        dupf[tid] = d;
    }
    __syncthreads();
    int lane = tid & 31, wave = tid >> 5;
    for (int c = wave; c < C_SZ; c += 8) {
        int j = cj[c];
        const float* col = WduT + (size_t)j * D_SZ;
        float s = 0.f;
        for (int d = lane; d < D_SZ; d += 32) s += wrow[d] * col[d];
        for (int o = 16; o > 0; o >>= 1) s += __shfl_down(s, o, 32);
        if (lane == 0 && !dupf[c]) {
            int pos = atomicAdd(&cursor[j], 1);
            col_fd[pos]  = fd;
            col_val[pos] = s;
        }
    }
}

// c_up = W_enc_down @ b_dec_up ; bias2 = b_enc_down - W_enc_down @ b_dec_down
__global__ __launch_bounds__(256)
void bias_kernel(const float* __restrict__ W_enc_down, const float* __restrict__ b_dec_up,
                 const float* __restrict__ b_dec_down, const float* __restrict__ b_enc_down,
                 float* __restrict__ c_up, float* __restrict__ bias2)
{
    int lane = threadIdx.x & 31, wave = threadIdx.x >> 5;
    int fd = blockIdx.x * 8 + wave;
    float s1 = 0.f, s2 = 0.f;
    for (int d = lane; d < D_SZ; d += 32) {
        float w = W_enc_down[(size_t)fd * D_SZ + d];
        s1 += w * b_dec_up[d];
        s2 += w * b_dec_down[d];
    }
    for (int o = 16; o > 0; o >>= 1) {
        s1 += __shfl_down(s1, o, 32);
        s2 += __shfl_down(s2, o, 32);
    }
    if (lane == 0) { c_up[fd] = s1; bias2[fd] = b_enc_down[fd] - s2; }
}

// ---------------------------------------------------------------------------
// Per-row top-64 of dense pre-activations (row resident in LDS)
// ---------------------------------------------------------------------------
__global__ __launch_bounds__(1024)
void topk_rows(const float* __restrict__ pre, float* __restrict__ vals,
               int* __restrict__ idx)
{
    __shared__ float row[FUP];
    __shared__ float rv[1024];
    __shared__ int   ri[1024];
    const int b = blockIdx.x, tid = threadIdx.x;
    for (int i = tid; i < FUP; i += 1024) row[i] = pre[(size_t)b * FUP + i];
    __syncthreads();
    for (int it = 0; it < K_TOP; ++it) {
        float best = -3.4e38f; int bi = 0;
        for (int i = tid; i < FUP; i += 1024) {
            float v = row[i];
            if (v > best) { best = v; bi = i; }
        }
        rv[tid] = best; ri[tid] = bi;
        __syncthreads();
        for (int s = 512; s > 0; s >>= 1) {
            if (tid < s && rv[tid + s] > rv[tid]) { rv[tid] = rv[tid + s]; ri[tid] = ri[tid + s]; }
            __syncthreads();
        }
        if (tid == 0) {
            vals[b * K_TOP + it] = rv[0];
            idx[b * K_TOP + it]  = ri[0];
            row[ri[0]] = -3.4e38f;
        }
        __syncthreads();
    }
}

// ---------------------------------------------------------------------------
// Fused per-row: sparse contributions (LDS atomics) -> scale/bias ->
// top-64 -> sparse decode to recon[b, 0:768]
// ---------------------------------------------------------------------------
__global__ __launch_bounds__(1024)
void fuse_rows(const float* __restrict__ P, const float* __restrict__ up_vals,
               const int* __restrict__ up_idx, const int* __restrict__ colptr,
               const int* __restrict__ col_fd, const float* __restrict__ col_val,
               const float* __restrict__ c_up, const float* __restrict__ bias2,
               const float* __restrict__ ln_scale, const float* __restrict__ WddT,
               const float* __restrict__ b_dec_down, float* __restrict__ out)
{
    __shared__ float row[FDOWN];
    __shared__ float rv[1024];
    __shared__ int   ri[1024];
    __shared__ float tv[K_TOP];
    __shared__ int   tix[K_TOP];
    const int b = blockIdx.x, tid = threadIdx.x;

    for (int i = tid; i < FDOWN; i += 1024) row[i] = P[(size_t)b * FDOWN + i];
    if (tid < K_TOP) { tv[tid] = up_vals[b * K_TOP + tid]; tix[tid] = up_idx[b * K_TOP + tid]; }
    __syncthreads();

    // sparse contributions: 64 active columns x ~256 CSC entries each
    for (int t = 0; t < K_TOP; ++t) {
        int j = tix[t]; float v = tv[t];
        int s = colptr[j], e = colptr[j + 1];
        for (int p = s + tid; p < e; p += 1024)
            atomicAdd(&row[col_fd[p]], v * col_val[p]);   // ds_add_f32
    }
    __syncthreads();

    float inv = 1.f / ln_scale[b];
    for (int i = tid; i < FDOWN; i += 1024)
        row[i] = (row[i] + c_up[i]) * inv + bias2[i];
    __syncthreads();

    // downstream top-64 (overwrites tv/tix)
    for (int it = 0; it < K_TOP; ++it) {
        float best = -3.4e38f; int bi = 0;
        for (int i = tid; i < FDOWN; i += 1024) {
            float v = row[i];
            if (v > best) { best = v; bi = i; }
        }
        rv[tid] = best; ri[tid] = bi;
        __syncthreads();
        for (int s = 512; s > 0; s >>= 1) {
            if (tid < s && rv[tid + s] > rv[tid]) { rv[tid] = rv[tid + s]; ri[tid] = ri[tid + s]; }
            __syncthreads();
        }
        if (tid == 0) { tv[it] = rv[0]; tix[it] = ri[0]; row[ri[0]] = -3.4e38f; }
        __syncthreads();
    }

    // sparse decode
    for (int d = tid; d < D_SZ; d += 1024) {
        float r = b_dec_down[d];
        for (int t = 0; t < K_TOP; ++t)
            r += tv[t] * WddT[(size_t)tix[t] * D_SZ + d];
        out[(size_t)b * D_SZ + d] = r;
    }
}

// ---------------------------------------------------------------------------
extern "C" void kernel_launch(void* const* d_in, const int* in_sizes, int n_in,
                              void* d_out, int out_size, void* d_ws, size_t ws_size,
                              hipStream_t stream)
{
    (void)in_sizes; (void)n_in; (void)out_size; (void)ws_size;
    const float* initial_acts = (const float*)d_in[0];
    const float* x_up         = (const float*)d_in[1];
    const float* ln_scale     = (const float*)d_in[2];
    const float* W_enc_up     = (const float*)d_in[3];
    const float* b_enc_up     = (const float*)d_in[4];
    const float* b_dec_up     = (const float*)d_in[5];
    const float* W_dec_up     = (const float*)d_in[6];
    const float* W_enc_down   = (const float*)d_in[7];
    const float* b_enc_down   = (const float*)d_in[8];
    const float* b_dec_down   = (const float*)d_in[9];
    const float* W_dec_down   = (const float*)d_in[10];
    const int*   connections  = (const int*)d_in[11];
    float*       out          = (float*)d_out;

    char* ws = (char*)d_ws;
    size_t off = 0;
    auto alloc = [&](size_t bytes) -> char* {
        char* p = ws + off;
        off += (bytes + 255) & ~(size_t)255;
        return p;
    };
    float*  P       = (float*) alloc((size_t)B_SZ * FUP * 4);     // reused for approx0
    float*  up_vals = (float*) alloc((size_t)B_SZ * K_TOP * 4);
    int*    up_idx  = (int*)   alloc((size_t)B_SZ * K_TOP * 4);
    float*  WduT    = (float*) alloc((size_t)FUP * D_SZ * 4);
    float*  WddT    = (float*) alloc((size_t)FDOWN * D_SZ * 4);
    int*    counts  = (int*)   alloc((size_t)FUP * 4);
    int*    colptr  = (int*)   alloc((size_t)(FUP + 1) * 4);
    int*    cursor  = (int*)   alloc((size_t)FUP * 4);
    int*    col_fd  = (int*)   alloc((size_t)FDOWN * C_SZ * 4);
    float*  col_val = (float*) alloc((size_t)FDOWN * C_SZ * 4);
    float*  c_up    = (float*) alloc((size_t)FDOWN * 4);
    float*  bias2   = (float*) alloc((size_t)FDOWN * 4);
    __bf16* Abf1    = (__bf16*)alloc((size_t)B_SZ * D_SZ * 2);    // bf16(x_up - b_dec_up)
    __bf16* Abf2    = (__bf16*)alloc((size_t)B_SZ * D_SZ * 2);    // bf16(initial_acts)
    __bf16* Wbf1    = (__bf16*)alloc((size_t)FUP * D_SZ * 2);     // bf16(W_enc_up)
    __bf16* Wbf2    = (__bf16*)alloc((size_t)FDOWN * D_SZ * 2);   // bf16(W_enc_down)

    // 0) bf16 operand preparation (subtract fused for the up-encoder input)
    {
        long long nA = (long long)B_SZ * D_SZ;
        long long nW = (long long)FUP * D_SZ;
        cvt_bf16<<<(unsigned)((nA + 255) / 256), 256, 0, stream>>>(x_up, b_dec_up, Abf1, D_SZ, nA);
        cvt_bf16<<<(unsigned)((nA + 255) / 256), 256, 0, stream>>>(initial_acts, nullptr, Abf2, D_SZ, nA);
        cvt_bf16<<<(unsigned)((nW + 255) / 256), 256, 0, stream>>>(W_enc_up, nullptr, Wbf1, D_SZ, nW);
        cvt_bf16<<<(unsigned)((nW + 255) / 256), 256, 0, stream>>>(W_enc_down, nullptr, Wbf2, D_SZ, nW);
    }

    // 1) transposes for contiguous sparse gathers
    transpose_kernel<<<dim3(FUP / 32, D_SZ / 32), dim3(32, 8), 0, stream>>>(W_dec_up, WduT, D_SZ, FUP);
    transpose_kernel<<<dim3(FDOWN / 32, D_SZ / 32), dim3(32, 8), 0, stream>>>(W_dec_down, WddT, D_SZ, FDOWN);

    // 2) CSC of masked virtual weights
    hipMemsetAsync(counts, 0, (size_t)FUP * 4, stream);
    count_kernel<<<FDOWN, C_SZ, 0, stream>>>(connections, counts);
    scan_kernel<<<1, 1024, 0, stream>>>(counts, colptr, cursor);
    fill_kernel<<<FDOWN, 256, 0, stream>>>(connections, W_enc_down, WduT, cursor, col_fd, col_val);

    // 3) bias GEMVs
    bias_kernel<<<FDOWN / 8, 256, 0, stream>>>(W_enc_down, b_dec_up, b_dec_down, b_enc_down, c_up, bias2);

    // 4) upstream encode (WMMA + TDM staging, fused ReLU + bias epilogue)
    gemm_nt_bf16<1><<<dim3(FUP / 128, B_SZ / 64), 256, 0, stream>>>(
        Abf1, Wbf1, b_enc_up, P, B_SZ, FUP, D_SZ);

    // 5) upstream top-64
    topk_rows<<<B_SZ, 1024, 0, stream>>>(P, up_vals, up_idx);

    // 6) initial contributions GEMM (reuses P)
    gemm_nt_bf16<0><<<dim3(FDOWN / 128, B_SZ / 64), 256, 0, stream>>>(
        Abf2, Wbf2, nullptr, P, B_SZ, FDOWN, D_SZ);

    // 7) fused sparse-contrib + scale/bias + top-64 + decode
    fuse_rows<<<B_SZ, 1024, 0, stream>>>(P, up_vals, up_idx, colptr, col_fd, col_val,
                                         c_up, bias2, ln_scale, WddT, b_dec_down, out);
}
